// STMoEFNN_20744692040184
// MI455X (gfx1250) — matrix operand order repaired and checked by reading, not compiled, not run on hardware
//
#include <hip/hip_runtime.h>
#include <hip/hip_bf16.h>
#include <math.h>
#include <stdint.h>

// ---- problem constants (match reference) ----
#define Bz   4
#define Nz   2048
#define Dz   512
#define Ez   8
#define Hz   1365
#define H2z  2730
#define CAPz 320      // max(4, min(2048, int(2048*1.25/8))) = 320
#define HPz  1376     // Hz padded to multiple of 32 (43*32) for guard-free GEMM2 K loop
#define BEz  (Bz*Ez)
#define MBz  5                // 64-row blocks over CAP (5*64 = 320)
#define TN1z 86               // ceil(1365/16) col tiles for H
#define TN2z (Dz/16)          // 32 col tiles for D
#define NC1z 6                // n-chunks of 16 tiles for GEMM1 (6*16 >= 86)
#define KCz  256              // K-chunk (halves) staged in LDS for GEMM2

typedef __attribute__((ext_vector_type(16))) _Float16 v16h;
typedef __attribute__((ext_vector_type(8)))  _Float16 v8h;
typedef __attribute__((ext_vector_type(8)))  float    v8f;

__device__ inline float rand01(unsigned int s) {
    s ^= s >> 17; s *= 0xed5ad4bbu;
    s ^= s >> 11; s *= 0xac4c1b51u;
    s ^= s >> 15; s *= 0x31848babu;
    s ^= s >> 14;
    return (float)(s >> 8) * (1.0f / 16777216.0f);
}

__device__ inline v16h cat16(v8h lo, v8h hi) {
    return __builtin_shufflevector(lo, hi, 0, 1, 2, 3, 4, 5, 6, 7, 8, 9, 10, 11, 12, 13, 14, 15);
}

// async 16B global -> LDS copy for the issuing lane (ASYNCcnt-tracked)
__device__ inline void async_g2l_b128(unsigned lds_off, const void* gptr) {
    unsigned long long ga = (unsigned long long)(uintptr_t)gptr;
    asm volatile("global_load_async_to_lds_b128 %0, %1, off"
                 :: "v"(lds_off), "v"(ga) : "memory");
}
__device__ inline void wait_async0() {
    asm volatile("s_wait_asynccnt 0x0" ::: "memory");
}

// ---------------- init scratch ----------------
__global__ void k_init(int* slot_token, float* dps, int* kept0, float* zsum) {
    int i = blockIdx.x * blockDim.x + threadIdx.x;
    if (i < Bz * Ez * CAPz) slot_token[i] = -1;
    if (i < Bz * Ez) { dps[i] = 0.0f; kept0[i] = 0; }
    if (i == 0) *zsum = 0.0f;
}

// ---------------- RMSNorm: xg = x/||x|| * sqrt(D) * gamma ----------------
__global__ __launch_bounds__(256) void k_rmsnorm(const float* __restrict__ x,
                                                 const float* __restrict__ gamma,
                                                 float* __restrict__ xg) {
    int row = blockIdx.x;                       // [0, B*N)
    __shared__ float red[256];
    const float* xr = x + (size_t)row * Dz;
    float s = 0.f;
    for (int d = threadIdx.x; d < Dz; d += 256) { float v = xr[d]; s += v * v; }
    red[threadIdx.x] = s; __syncthreads();
    for (int o = 128; o > 0; o >>= 1) {
        if (threadIdx.x < o) red[threadIdx.x] += red[threadIdx.x + o];
        __syncthreads();
    }
    float nrm = sqrtf(red[0]);
    float scale = 22.62741699796952f / fmaxf(nrm, 1e-12f);   // sqrt(512)
    for (int d = threadIdx.x; d < Dz; d += 256)
        xg[(size_t)row * Dz + d] = xr[d] * scale * gamma[d];
}

// ---------------- w1 [E][D][2H] f32 -> w1t [E][2H][D] f16 (transposed) ----------------
__global__ void k_w1t(const float* __restrict__ w1, _Float16* __restrict__ w1t) {
    size_t n = (size_t)Ez * H2z * Dz;
    size_t i = (size_t)blockIdx.x * blockDim.x + threadIdx.x;
    size_t st = (size_t)gridDim.x * blockDim.x;
    for (; i < n; i += st) {
        int k   = (int)(i % Dz);
        int col = (int)((i / Dz) % H2z);
        int e   = (int)(i / ((size_t)Dz * H2z));
        w1t[i] = (_Float16)w1[((size_t)e * Dz + k) * H2z + col];
    }
}

// ---------------- w2 [E][H][D] f32 -> w2t [E][D][HP] f16 (transposed, K zero-padded) ----
__global__ void k_w2t(const float* __restrict__ w2, _Float16* __restrict__ w2t) {
    size_t n = (size_t)Ez * Dz * HPz;
    size_t i = (size_t)blockIdx.x * blockDim.x + threadIdx.x;
    size_t st = (size_t)gridDim.x * blockDim.x;
    for (; i < n; i += st) {
        int k = (int)(i % HPz);
        int d = (int)((i / HPz) % Dz);
        int e = (int)(i / ((size_t)HPz * Dz));
        w2t[i] = (k < Hz) ? (_Float16)w2[((size_t)e * Hz + k) * Dz + d] : (_Float16)0.f;
    }
}

// ---------------- gating: one wave per token ----------------
__global__ __launch_bounds__(32) void k_gating(const float* __restrict__ xg,
                                               const float* __restrict__ gate_w,
                                               int* e0a, int* e1a, float* g0a, float* g1a, int* r1a,
                                               float* dps, float* zsum) {
    int t = blockIdx.x;                          // token in [0, B*N)
    int lane = threadIdx.x;
    const float* xr = xg + (size_t)t * Dz;
    float acc[Ez];
    #pragma unroll
    for (int e = 0; e < Ez; e++) acc[e] = 0.f;
    for (int d = lane; d < Dz; d += 32) {
        float xv = xr[d];
        #pragma unroll
        for (int e = 0; e < Ez; e++) acc[e] += xv * gate_w[d * Ez + e];
    }
    #pragma unroll
    for (int off = 16; off > 0; off >>= 1) {
        #pragma unroll
        for (int e = 0; e < Ez; e++) acc[e] += __shfl_down(acc[e], off, 32);
    }
    if (lane == 0) {
        float mx = acc[0];
        #pragma unroll
        for (int e = 1; e < Ez; e++) mx = fmaxf(mx, acc[e]);
        float p[Ez]; float s = 0.f;
        #pragma unroll
        for (int e = 0; e < Ez; e++) { p[e] = __expf(acc[e] - mx); s += p[e]; }
        float inv = 1.f / s;
        #pragma unroll
        for (int e = 0; e < Ez; e++) p[e] *= inv;
        int b = t / Nz;
        #pragma unroll
        for (int e = 0; e < Ez; e++) atomicAdd(&dps[b * Ez + e], p[e]);
        float z = __logf(s) + mx;
        atomicAdd(zsum, z * z);
        int e0 = 0;
        #pragma unroll
        for (int e = 1; e < Ez; e++) if (p[e] > p[e0]) e0 = e;
        int e1 = (e0 == 0) ? 1 : 0;
        #pragma unroll
        for (int e = 0; e < Ez; e++) if (e != e0 && p[e] > p[e1]) e1 = e;
        float g0 = p[e0], g1 = p[e1];
        float den = fmaxf(g0 + g1, 1e-9f);
        float g0n = g0 / den, g1n = g1 / den;
        float u = rand01(0x9E3779B9u ^ ((unsigned)t * 2654435761u + 12345u));
        int r1 = (u < g1n * 5.0f) ? 1 : 0;       // prob = min(1, gate/0.2)
        e0a[t] = e0; e1a[t] = e1; g0a[t] = g0n; g1a[t] = g1n; r1a[t] = r1;
    }
}

// ---------------- capacity routing (serial exclusive-cumsum semantics) ----------------
__global__ void k_route(const int* e0a, const int* e1a, const float* g0a, const float* g1a,
                        const int* r1a, int* slot_token, int* ape, int* app, float* apg,
                        int* kept0) {
    int b = blockIdx.x;
    if (threadIdx.x != 0) return;
    int cnt[Ez], prev[Ez], k0c[Ez];
    for (int e = 0; e < Ez; e++) { cnt[e] = 0; k0c[e] = 0; }
    // k = 0 (always routed)
    for (int n = 0; n < Nz; n++) {
        int t = b * Nz + n;
        int e = e0a[t];
        int pos = cnt[e]++;                      // cumsum of mask (pre-capacity)
        int idx = (0 * Bz + b) * Nz + n;
        if (pos < CAPz) {
            slot_token[(b * Ez + e) * CAPz + pos] = n;
            ape[idx] = e; app[idx] = pos; apg[idx] = g0a[t];
            k0c[e]++;
        } else app[idx] = -1;
    }
    for (int e = 0; e < Ez; e++) { prev[e] = k0c[e]; cnt[e] = 0; kept0[b * Ez + e] = k0c[e]; }
    // k = 1 (stochastic)
    for (int n = 0; n < Nz; n++) {
        int t = b * Nz + n;
        int idx = (1 * Bz + b) * Nz + n;
        if (!r1a[t]) { app[idx] = -1; continue; }
        int e = e1a[t];
        int pos = cnt[e]++ + prev[e];
        if (pos < CAPz) {
            slot_token[(b * Ez + e) * CAPz + pos] = n;
            ape[idx] = e; app[idx] = pos; apg[idx] = g1a[t];
        } else app[idx] = -1;
    }
}

// ---------------- dispatch gather: expert_in[b,e,c,:] (f16) ----------------
__global__ __launch_bounds__(128) void k_dispatch(const int* __restrict__ slot_token,
                                                  const float* __restrict__ xg,
                                                  _Float16* __restrict__ ein) {
    int slot = blockIdx.x;                       // [0, B*E*CAP)
    int be = slot / CAPz;
    int b = be / Ez;
    int tok = slot_token[slot];
    _Float16* dst = ein + (size_t)slot * Dz;
    if (tok >= 0) {
        const float* src = xg + ((size_t)b * Nz + tok) * Dz;
        for (int d = threadIdx.x; d < Dz; d += 128) dst[d] = (_Float16)src[d];
    } else {
        for (int d = threadIdx.x; d < Dz; d += 128) dst[d] = (_Float16)0.f;
    }
}

// ---------------- GEMM1 + GEGLU, async-LDS-staged A ----------------
// Workgroup = 8 waves. A panel 64x512 f16 (64 KB) staged in LDS once via
// global_load_async_to_lds_b128; each wave computes 2 N-tiles (u+g, 4 M-tiles,
// 8 WMMA / k-step).
__global__ __launch_bounds__(256) void k_gemm1(const _Float16* __restrict__ ein,
                                               const _Float16* __restrict__ w1t,
                                               const float* __restrict__ b1,
                                               const float* __restrict__ mult_bias,
                                               _Float16* __restrict__ act) {
    __shared__ _Float16 As[64 * Dz];             // 64 KB
    int bid = blockIdx.x;
    int nc = bid % NC1z;
    int mb = (bid / NC1z) % MBz;
    int be = bid / (NC1z * MBz);
    int e  = be % Ez;
    int tid   = threadIdx.x;
    int w     = tid >> 5;
    int lane  = tid & 31;
    int mrow  = lane & 15;
    int khalf = lane >> 4;
    int ncol  = lane & 15;
    int kboff = khalf * 16;

    // stage contiguous 64x512 A panel (ein rows are contiguous) via async copies
    const _Float16* Ablk = ein + (size_t)be * CAPz * Dz + (size_t)(mb * 64) * Dz;
    {
        unsigned ldsbase = (unsigned)(uintptr_t)As;      // low 32 bits = LDS offset
        #pragma unroll
        for (int c = 0; c < 16; ++c) {
            int i = c * 256 + tid;                        // 16-byte unit index
            async_g2l_b128(ldsbase + (unsigned)i * 16, (const char*)Ablk + (size_t)i * 16);
        }
        wait_async0();
    }
    __syncthreads();

    _Float16* actb = act + (size_t)be * CAPz * HPz + (size_t)(mb * 64) * HPz;
    for (int rep = 0; rep < 2; ++rep) {
        int tn = nc * 16 + rep * 8 + w;
        if (tn >= TN1z) continue;
        int colu = tn * 16 + ncol;                               // < 1376 < 2730: valid row
        int colg = Hz + tn * 16 + ncol;
        colg = (colg < H2z) ? colg : (H2z - 1);                  // clamp; garbage never stored
        const _Float16* Bu = w1t + ((size_t)e * H2z + colu) * Dz;
        const _Float16* Bg = w1t + ((size_t)e * H2z + colg) * Dz;

        v8f cu[4] = {}; v8f cg[4] = {};
        for (int k0 = 0; k0 < Dz; k0 += 32) {
            v16h bu = *(const v16h*)(Bu + k0 + kboff);
            v16h bg = *(const v16h*)(Bg + k0 + kboff);
            __builtin_prefetch(Bu + k0 + 128, 0, 1);
            #pragma unroll
            for (int t = 0; t < 4; t++) {
                const _Float16* ap = As + (t * 16 + mrow) * Dz + k0 + khalf * 16;
                v16h a = *(const v16h*)ap;
                cu[t] = __builtin_amdgcn_wmma_f32_16x16x32_f16(false, a, false, bu, (short)0, cu[t], false, false);
                cg[t] = __builtin_amdgcn_wmma_f32_16x16x32_f16(false, a, false, bg, (short)0, cg[t], false, false);
            }
        }
        int col = tn * 16 + ncol;
        if (col < Hz) {
            float mbv = mult_bias[e * Hz + col];
            float bb0 = b1[e * H2z + col];
            float bb1 = b1[e * H2z + Hz + col];
            #pragma unroll
            for (int t = 0; t < 4; t++) {
                #pragma unroll
                for (int r = 0; r < 8; ++r) {
                    int m = t * 16 + khalf * 8 + r;
                    float u = cu[t][r] + bb0;
                    float g = cg[t][r] + bb1;
                    float ge = 0.5f * g * (1.0f + erff(g * 0.7071067811865475f));  // exact gelu
                    actb[(size_t)m * HPz + col] = (_Float16)(u * ge * mbv);
                }
            }
        } else {                                                  // zero K-pad region
            #pragma unroll
            for (int t = 0; t < 4; t++) {
                #pragma unroll
                for (int r = 0; r < 8; ++r) {
                    int m = t * 16 + khalf * 8 + r;
                    actb[(size_t)m * HPz + col] = (_Float16)0.f;
                }
            }
        }
    }
}

// ---------------- GEMM2, async-LDS-staged A with K-chunking ----------------
// Workgroup = 8 waves; A chunk 64x256 halves (32 KB) async-staged per K-chunk.
// Each wave: 2 reps x 2 N-tiles x 4 M-tiles (8 WMMA / k-step).
__global__ __launch_bounds__(256) void k_gemm2(const _Float16* __restrict__ act,
                                               const _Float16* __restrict__ w2t,
                                               const float* __restrict__ b2,
                                               float* __restrict__ eout) {
    __shared__ _Float16 As[64 * KCz];            // 32 KB
    int bid = blockIdx.x;                        // [0, BE*MB)
    int mb = bid % MBz;
    int be = bid / MBz;
    int e  = be % Ez;
    int tid   = threadIdx.x;
    int w     = tid >> 5;
    int lane  = tid & 31;
    int mrow  = lane & 15;
    int khalf = lane >> 4;
    int ncol  = lane & 15;
    int kboff = khalf * 16;

    const _Float16* Ablk = act + (size_t)be * CAPz * HPz + (size_t)(mb * 64) * HPz;
    float* outb = eout + (size_t)be * CAPz * Dz + (size_t)(mb * 64) * Dz;
    unsigned ldsbase = (unsigned)(uintptr_t)As;

    for (int rep = 0; rep < 2; ++rep) {
        int tnA = rep * 16 + w;                  // two N-tiles per wave per rep
        int tnB = rep * 16 + 8 + w;
        int colA = tnA * 16 + ncol;
        int colB = tnB * 16 + ncol;
        const _Float16* BpA = w2t + ((size_t)e * Dz + colA) * HPz;
        const _Float16* BpB = w2t + ((size_t)e * Dz + colB) * HPz;
        v8f cA[4] = {}; v8f cB[4] = {};

        for (int kc = 0; kc < 6; ++kc) {         // 6 chunks: 5*256 + 96 = 1376
            // async-stage 64 x 256 chunk (rows strided by HPz)
            #pragma unroll
            for (int c = 0; c < 8; ++c) {
                int i = c * 256 + tid;           // 16-byte unit index within chunk
                int row = i >> 5;                // 256 halves = 32 units per row
                int kk = (i & 31) * 8;
                async_g2l_b128(ldsbase + (unsigned)i * 16,
                               Ablk + (size_t)row * HPz + kc * KCz + kk);
            }
            wait_async0();
            __syncthreads();
            int ksteps = (kc < 5) ? 8 : 3;       // last chunk covers 96 halves
            for (int ks = 0; ks < ksteps; ++ks) {
                int kl = ks * 32;
                int kg = kc * KCz + kl;
                v16h bA = *(const v16h*)(BpA + kg + kboff);
                v16h bB = *(const v16h*)(BpB + kg + kboff);
                __builtin_prefetch(BpA + kg + 128, 0, 1);
                #pragma unroll
                for (int t = 0; t < 4; t++) {
                    const _Float16* ap = As + (t * 16 + mrow) * KCz + kl + khalf * 16;
                    v16h a = *(const v16h*)ap;
                    cA[t] = __builtin_amdgcn_wmma_f32_16x16x32_f16(false, a, false, bA, (short)0, cA[t], false, false);
                    cB[t] = __builtin_amdgcn_wmma_f32_16x16x32_f16(false, a, false, bB, (short)0, cB[t], false, false);
                }
            }
            __syncthreads();
        }
        float bbA = b2[e * Dz + colA];
        float bbB = b2[e * Dz + colB];
        #pragma unroll
        for (int t = 0; t < 4; t++) {
            #pragma unroll
            for (int r = 0; r < 8; ++r) {
                int m = t * 16 + khalf * 8 + r;
                outb[(size_t)m * Dz + colA] = cA[t][r] + bbA;
                outb[(size_t)m * Dz + colB] = cB[t][r] + bbB;
            }
        }
    }
}

// ---------------- combine + residual + LayerNorm ----------------
__global__ __launch_bounds__(256) void k_combine_ln(const float* __restrict__ x,
                                                    const float* __restrict__ eout,
                                                    const int* __restrict__ ape,
                                                    const int* __restrict__ app,
                                                    const float* __restrict__ apg,
                                                    const float* __restrict__ ln_g,
                                                    const float* __restrict__ ln_b,
                                                    float* __restrict__ out) {
    int row = blockIdx.x;                        // b*N + n
    int b = row / Nz, n = row % Nz;
    __shared__ float red[256];
    float y[2];
    #pragma unroll
    for (int i = 0; i < 2; i++) {
        int d = threadIdx.x + i * 256;
        float acc = x[(size_t)row * Dz + d];
        #pragma unroll
        for (int k = 0; k < 2; k++) {
            int idx = (k * Bz + b) * Nz + n;
            int p = app[idx];
            if (p >= 0) {
                int e = ape[idx];
                acc += apg[idx] * eout[(((size_t)(b * Ez + e)) * CAPz + p) * Dz + d];
            }
        }
        y[i] = acc;
    }
    red[threadIdx.x] = y[0] + y[1]; __syncthreads();
    for (int o = 128; o > 0; o >>= 1) {
        if (threadIdx.x < o) red[threadIdx.x] += red[threadIdx.x + o];
        __syncthreads();
    }
    float mu = red[0] / Dz;
    __syncthreads();
    red[threadIdx.x] = (y[0] - mu) * (y[0] - mu) + (y[1] - mu) * (y[1] - mu);
    __syncthreads();
    for (int o = 128; o > 0; o >>= 1) {
        if (threadIdx.x < o) red[threadIdx.x] += red[threadIdx.x + o];
        __syncthreads();
    }
    float rstd = rsqrtf(red[0] / Dz + 1e-5f);
    #pragma unroll
    for (int i = 0; i < 2; i++) {
        int d = threadIdx.x + i * 256;
        out[(size_t)row * Dz + d] = (y[i] - mu) * rstd * ln_g[d] + ln_b[d];
    }
}

// ---------------- aux loss scalar ----------------
__global__ void k_finalize(const float* dps, const int* kept0, const float* zsum, float* out_scalar) {
    if (threadIdx.x || blockIdx.x) return;
    float s = 0.f;
    for (int i = 0; i < Bz * Ez; i++)
        s += (dps[i] / (float)Nz) * ((float)kept0[i] / (float)Nz);
    float bal = s / (float)(Bz * Ez) * (float)(Ez * Ez);
    float zl = (*zsum) / (float)(Bz * Nz);
    *out_scalar = 0.01f * bal + 0.001f * zl;
}

extern "C" void kernel_launch(void* const* d_in, const int* in_sizes, int n_in,
                              void* d_out, int out_size, void* d_ws, size_t ws_size,
                              hipStream_t stream) {
    const float* x         = (const float*)d_in[0];
    const float* gate_w    = (const float*)d_in[1];
    const float* prenorm_g = (const float*)d_in[2];
    const float* w1        = (const float*)d_in[3];
    const float* b1        = (const float*)d_in[4];
    const float* mult_bias = (const float*)d_in[5];
    const float* w2        = (const float*)d_in[6];
    const float* b2        = (const float*)d_in[7];
    const float* ln_g      = (const float*)d_in[8];
    const float* ln_b      = (const float*)d_in[9];
    float* out = (float*)d_out;

    char* ws = (char*)d_ws;
    size_t off = 0;
    auto alloc = [&](size_t bytes) -> char* {
        char* p = ws + off;
        off = (off + bytes + 255) & ~(size_t)255;
        return p;
    };
    float*    xg   = (float*)   alloc((size_t)Bz * Nz * Dz * 4);
    _Float16* w1t  = (_Float16*)alloc((size_t)Ez * H2z * Dz * 2);   // [E][2H][D]
    _Float16* w2t  = (_Float16*)alloc((size_t)Ez * Dz * HPz * 2);   // [E][D][HP]
    _Float16* ein  = (_Float16*)alloc((size_t)Bz * Ez * CAPz * Dz * 2);
    _Float16* act  = (_Float16*)alloc((size_t)Bz * Ez * CAPz * HPz * 2);
    float*    eout = (float*)   alloc((size_t)Bz * Ez * CAPz * Dz * 4);
    int*      e0a  = (int*)     alloc((size_t)Bz * Nz * 4);
    int*      e1a  = (int*)     alloc((size_t)Bz * Nz * 4);
    float*    g0a  = (float*)   alloc((size_t)Bz * Nz * 4);
    float*    g1a  = (float*)   alloc((size_t)Bz * Nz * 4);
    int*      r1a  = (int*)     alloc((size_t)Bz * Nz * 4);
    int*      ape  = (int*)     alloc((size_t)2 * Bz * Nz * 4);
    int*      app  = (int*)     alloc((size_t)2 * Bz * Nz * 4);
    float*    apg  = (float*)   alloc((size_t)2 * Bz * Nz * 4);
    int*      slot = (int*)     alloc((size_t)Bz * Ez * CAPz * 4);
    float*    dps  = (float*)   alloc((size_t)Bz * Ez * 4);
    int*      kp0  = (int*)     alloc((size_t)Bz * Ez * 4);
    float*    zsum = (float*)   alloc(4);
    (void)ws_size; (void)in_sizes; (void)n_in; (void)out_size;

    k_init<<<(Bz * Ez * CAPz + 255) / 256, 256, 0, stream>>>(slot, dps, kp0, zsum);
    k_rmsnorm<<<Bz * Nz, 256, 0, stream>>>(x, prenorm_g, xg);
    k_w1t<<<4096, 256, 0, stream>>>(w1, w1t);
    k_w2t<<<4096, 256, 0, stream>>>(w2, w2t);
    k_gating<<<Bz * Nz, 32, 0, stream>>>(xg, gate_w, e0a, e1a, g0a, g1a, r1a, dps, zsum);
    k_route<<<Bz, 32, 0, stream>>>(e0a, e1a, g0a, g1a, r1a, slot, ape, app, apg, kp0);
    k_dispatch<<<Bz * Ez * CAPz, 128, 0, stream>>>(slot, xg, ein);
    k_gemm1<<<BEz * MBz * NC1z, 256, 0, stream>>>(ein, w1t, b1, mult_bias, act);
    k_gemm2<<<BEz * MBz, 256, 0, stream>>>(act, w2t, b2, eout);
    k_combine_ln<<<Bz * Nz, 256, 0, stream>>>(x, eout, ape, app, apg, ln_g, ln_b, out);
    k_finalize<<<1, 1, 0, stream>>>(dps, kp0, zsum, out + (size_t)Bz * Nz * Dz);
}
